// RotarySelfAttention_2946347565435
// MI455X (gfx1250) — matrix operand order repaired and checked by reading, not compiled, not run on hardware
//
#include <hip/hip_runtime.h>

#define BATCH  4
#define SLEN   2048
#define DMODEL 1024
#define NHEAD  16
#define HDIM   64
#define PSTRIDE 40   // LDS row stride in floats (160B, 16B aligned)

typedef __attribute__((ext_vector_type(16))) __bf16 v16bf;
typedef __attribute__((ext_vector_type(8)))  __bf16 v8bf;
typedef __attribute__((ext_vector_type(8)))  float  v8f;
typedef __attribute__((ext_vector_type(4)))  float  v4f;
typedef __attribute__((ext_vector_type(16))) unsigned char uc16;

static __device__ inline __bf16 f2bf(float f) {
  unsigned u = __builtin_bit_cast(unsigned, f);
  unsigned r = (u + 0x7FFFu + ((u >> 16) & 1u)) >> 16;   // RNE
  unsigned short s = (unsigned short)r;
  return __builtin_bit_cast(__bf16, s);
}

// Assemble a 16-element bf16 WMMA fragment from two contiguous 8-element (16B) chunks.
static __device__ inline v16bf ld_frag(const __bf16* p0, const __bf16* p1) {
  v8bf lo = *reinterpret_cast<const v8bf*>(p0);
  v8bf hi = *reinterpret_cast<const v8bf*>(p1);
  v16bf r;
#pragma unroll
  for (int i = 0; i < 8; ++i) { r[i] = lo[i]; r[i + 8] = hi[i]; }
  return r;
}

__global__ __launch_bounds__(256) void convert_bf16(const float* __restrict__ src,
                                                    __bf16* __restrict__ dst, int n) {
  int i = blockIdx.x * 256 + threadIdx.x;
  if (i < n) dst[i] = f2bf(src[i]);
}

// ---------------------------------------------------------------------------
// Kernel 1: qkv = x @ w_qkv^T (bf16 WMMA, f32 accum), RoPE fused in epilogue.
// Wave tile: 32 (rows of B*S) x 64 (cols of 3D): 8 WMMAs per 12 b128 loads.
// A 64-col band is exactly one q/k/v segment of one head. Writes Q,K (roped,
// Q pre-scaled by 1/8) as [b,h,s,hd] bf16, V transposed as [b,h,hd,s] bf16.
// ---------------------------------------------------------------------------
__global__ __launch_bounds__(256) void qkv_rope(const __bf16* __restrict__ xb,
                                                const __bf16* __restrict__ wb,
                                                __bf16* __restrict__ Qb,
                                                __bf16* __restrict__ Kb,
                                                __bf16* __restrict__ Vtb) {
  const int lane = threadIdx.x & 31;
  const int wave = threadIdx.x >> 5;
  const int l15  = lane & 15;
  const int hl   = lane >> 4;
  const int n_band = blockIdx.x;                    // 0..47
  const int m0 = (blockIdx.y * 8 + wave) * 32;      // rows m0..m0+31

  const __bf16* aRow0 = xb + (size_t)(m0 + l15) * DMODEL;
  const __bf16* aRow1 = xb + (size_t)(m0 + 16 + l15) * DMODEL;
  const __bf16* bRow[4];
#pragma unroll
  for (int t = 0; t < 4; ++t)
    bRow[t] = wb + (size_t)(n_band * 64 + t * 16 + l15) * DMODEL;

  v8f acc[2][4];
#pragma unroll
  for (int mi = 0; mi < 2; ++mi)
#pragma unroll
    for (int t = 0; t < 4; ++t)
#pragma unroll
      for (int i = 0; i < 8; ++i) acc[mi][t][i] = 0.0f;

  for (int k = 0; k < DMODEL; k += 32) {
    v16bf A0 = ld_frag(aRow0 + k + hl * 8, aRow0 + k + hl * 8 + 16);
    v16bf A1 = ld_frag(aRow1 + k + hl * 8, aRow1 + k + hl * 8 + 16);
#pragma unroll
    for (int t = 0; t < 4; ++t) {
      v16bf Bf = ld_frag(bRow[t] + k + hl * 8, bRow[t] + k + hl * 8 + 16);
      acc[0][t] = __builtin_amdgcn_wmma_f32_16x16x32_bf16(false, A0, false, Bf,
                                                          (short)0, acc[0][t], false, false);
      acc[1][t] = __builtin_amdgcn_wmma_f32_16x16x32_bf16(false, A1, false, Bf,
                                                          (short)0, acc[1][t], false, false);
    }
  }

  const int bb  = m0 / SLEN;                 // m0 is a multiple of 32, S mult of 32
  const int h   = n_band / 3;
  const int seg = n_band % 3;                // 0=q 1=k 2=v
  const int bh  = bb * NHEAD + h;

#pragma unroll
  for (int mi = 0; mi < 2; ++mi) {
    const int s_base = (m0 % SLEN) + mi * 16 + 8 * hl;
    if (seg == 2) {
      // V: store transposed [b,h,hd,s]; per lane rows v are consecutive s.
#pragma unroll
      for (int t = 0; t < 4; ++t) {
        const int d = t * 16 + l15;
        v8bf pk;
#pragma unroll
        for (int v = 0; v < 8; ++v) pk[v] = f2bf(acc[mi][t][v]);
        *reinterpret_cast<v8bf*>(Vtb + ((size_t)bh * HDIM + d) * SLEN + s_base) = pk;
      }
    } else {
      __bf16* dst = (seg == 0) ? Qb : Kb;
      const float qs = (seg == 0) ? 0.125f : 1.0f;   // fold 1/sqrt(HD) into Q
#pragma unroll
      for (int t = 0; t < 2; ++t) {                  // tile t pairs with t+2
        const int j = t * 16 + l15;                  // 0..31
        const float inv = __expf(-(float)j * 0.2878225386f);  // 10000^(-j/32)
#pragma unroll
        for (int v = 0; v < 8; ++v) {
          const int s = s_base + v;
          float sn, cs;
          __sincosf((float)s * inv, &sn, &cs);
          const float x1 = acc[mi][t][v];
          const float x2 = acc[mi][t + 2][v];
          __bf16* p = dst + ((size_t)bh * SLEN + s) * HDIM;
          p[j]      = f2bf((x1 * cs - x2 * sn) * qs);
          p[j + 32] = f2bf((x2 * cs + x1 * sn) * qs);
        }
      }
    }
  }
}

// ---------------------------------------------------------------------------
// Kernel 2: flash attention. One wave = 16 queries of one (b,h); iterates over
// keys in chunks of 32. Logits via bf16 WMMA, online softmax with per-wave LDS
// staging (C-layout -> Pt B-operand layout). Mask is applied row-side after
// the staging (one aligned 16-byte load per lane). O accumulated transposed
// (Ot = Vt x Pt) so the softmax rescale is a per-lane scalar.
// ---------------------------------------------------------------------------
__global__ __launch_bounds__(256) void flash_attn(const __bf16* __restrict__ Qb,
                                                  const __bf16* __restrict__ Kb,
                                                  const __bf16* __restrict__ Vtb,
                                                  const unsigned char* __restrict__ mask,
                                                  __bf16* __restrict__ valsb) {
  __shared__ float ldsP[8][16 * PSTRIDE];
  const int lane = threadIdx.x & 31;
  const int wave = threadIdx.x >> 5;
  const int l15  = lane & 15;
  const int hl   = lane >> 4;
  const int bh  = blockIdx.x;                 // 0..63
  const int b   = bh >> 4;
  const int h   = bh & 15;
  const int sq0 = blockIdx.y * 128 + wave * 16;

  const __bf16* qrow = Qb + ((size_t)bh * SLEN + sq0 + l15) * HDIM;
  v16bf Qf[2];
#pragma unroll
  for (int f = 0; f < 2; ++f)
    Qf[f] = ld_frag(qrow + f * 32 + hl * 16, qrow + f * 32 + hl * 16 + 8);

  v8f O[4];
#pragma unroll
  for (int t = 0; t < 4; ++t)
#pragma unroll
    for (int i = 0; i < 8; ++i) O[t][i] = 0.0f;

  float m_run = -3.0e38f;
  float l_run = 0.0f;

  // lane-owned mask row: q = sq0 + l15, 16 keys starting at c0 + hl*16
  const unsigned char* mrow = mask + (size_t)b * SLEN * SLEN
                                   + (size_t)(sq0 + l15) * SLEN + hl * 16;
  float* myP = &ldsP[wave][0];

  for (int c0 = 0; c0 < SLEN; c0 += 32) {
#pragma unroll
    for (int kt = 0; kt < 2; ++kt) {
      const int kcol = c0 + kt * 16 + l15;
      const __bf16* krow = Kb + ((size_t)bh * SLEN + kcol) * HDIM;
      v16bf B0 = ld_frag(krow + hl * 16,      krow + hl * 16 + 8);
      v16bf B1 = ld_frag(krow + 32 + hl * 16, krow + 32 + hl * 16 + 8);
      v8f sacc;
#pragma unroll
      for (int i = 0; i < 8; ++i) sacc[i] = 0.0f;
      sacc = __builtin_amdgcn_wmma_f32_16x16x32_bf16(false, Qf[0], false, B0,
                                                     (short)0, sacc, false, false);
      sacc = __builtin_amdgcn_wmma_f32_16x16x32_bf16(false, Qf[1], false, B1,
                                                     (short)0, sacc, false, false);
      // stage raw logits to LDS (row-major 16q x 32k tile per wave)
#pragma unroll
      for (int v = 0; v < 8; ++v)
        myP[(v + 8 * hl) * PSTRIDE + kt * 16 + l15] = sacc[v];
    }
    asm volatile("s_wait_dscnt 0" ::: "memory");

    // each lane re-reads row q=l15, keys hl*16..hl*16+15 (== Pt B-operand slot)
    const float* prow = myP + l15 * PSTRIDE + hl * 16;
    float x[16];
    {
      v4f r0 = *reinterpret_cast<const v4f*>(prow);
      v4f r1 = *reinterpret_cast<const v4f*>(prow + 4);
      v4f r2 = *reinterpret_cast<const v4f*>(prow + 8);
      v4f r3 = *reinterpret_cast<const v4f*>(prow + 12);
#pragma unroll
      for (int i = 0; i < 4; ++i) { x[i] = r0[i]; x[4+i] = r1[i]; x[8+i] = r2[i]; x[12+i] = r3[i]; }
    }
    // apply mask: one aligned 16-byte load covers this lane's 16 keys
    const uc16 mv = *reinterpret_cast<const uc16*>(mrow + c0);
#pragma unroll
    for (int i = 0; i < 16; ++i)
      if (mv[i]) x[i] = -9.0e15f;

    float mt = x[0];
#pragma unroll
    for (int i = 1; i < 16; ++i) mt = fmaxf(mt, x[i]);
    mt = fmaxf(mt, __shfl_xor(mt, 16, 32));          // buddy lane covers other 16 keys
    const float m_new = fmaxf(m_run, mt);
    const float alpha = __expf(m_run - m_new);
    v16bf Pf;
    float ssum = 0.0f;
#pragma unroll
    for (int i = 0; i < 16; ++i) {
      const float p = __expf(x[i] - m_new);
      ssum += p;
      Pf[i] = f2bf(p);
    }
    ssum += __shfl_xor(ssum, 16, 32);
    l_run = l_run * alpha + ssum;
    m_run = m_new;
#pragma unroll
    for (int t = 0; t < 4; ++t) O[t] *= alpha;       // per-lane scalar (col = query)
#pragma unroll
    for (int t = 0; t < 4; ++t) {
      const __bf16* vrow = Vtb + ((size_t)bh * HDIM + t * 16 + l15) * SLEN + c0;
      v16bf Vf = ld_frag(vrow + hl * 16, vrow + hl * 16 + 8);   // A-frag of V^T
      O[t] = __builtin_amdgcn_wmma_f32_16x16x32_bf16(false, Vf, false, Pf,
                                                     (short)0, O[t], false, false);
    }
  }

  const float invl = 1.0f / l_run;
  const int q = sq0 + l15;
#pragma unroll
  for (int t = 0; t < 4; ++t) {                      // Ot tile rows d consecutive
    v8bf pk;
#pragma unroll
    for (int v = 0; v < 8; ++v) pk[v] = f2bf(O[t][v] * invl);
    __bf16* dst = valsb + (size_t)(b * SLEN + q) * DMODEL + h * HDIM + t * 16 + hl * 8;
    *reinterpret_cast<v8bf*>(dst) = pk;
  }
}

// ---------------------------------------------------------------------------
// Kernel 3: out = vals @ w_out^T, f32 output. 32x64 wave tile like kernel 1.
// ---------------------------------------------------------------------------
__global__ __launch_bounds__(256) void out_proj(const __bf16* __restrict__ ab,
                                                const __bf16* __restrict__ wb,
                                                float* __restrict__ out) {
  const int lane = threadIdx.x & 31;
  const int wave = threadIdx.x >> 5;
  const int l15  = lane & 15;
  const int hl   = lane >> 4;
  const int n_band = blockIdx.x;                    // 0..15
  const int m0 = (blockIdx.y * 8 + wave) * 32;      // rows m0..m0+31

  const __bf16* aRow0 = ab + (size_t)(m0 + l15) * DMODEL;
  const __bf16* aRow1 = ab + (size_t)(m0 + 16 + l15) * DMODEL;
  const __bf16* bRow[4];
#pragma unroll
  for (int t = 0; t < 4; ++t)
    bRow[t] = wb + (size_t)(n_band * 64 + t * 16 + l15) * DMODEL;

  v8f acc[2][4];
#pragma unroll
  for (int mi = 0; mi < 2; ++mi)
#pragma unroll
    for (int t = 0; t < 4; ++t)
#pragma unroll
      for (int i = 0; i < 8; ++i) acc[mi][t][i] = 0.0f;

  for (int k = 0; k < DMODEL; k += 32) {
    v16bf A0 = ld_frag(aRow0 + k + hl * 8, aRow0 + k + hl * 8 + 16);
    v16bf A1 = ld_frag(aRow1 + k + hl * 8, aRow1 + k + hl * 8 + 16);
#pragma unroll
    for (int t = 0; t < 4; ++t) {
      v16bf Bf = ld_frag(bRow[t] + k + hl * 8, bRow[t] + k + hl * 8 + 16);
      acc[0][t] = __builtin_amdgcn_wmma_f32_16x16x32_bf16(false, A0, false, Bf,
                                                          (short)0, acc[0][t], false, false);
      acc[1][t] = __builtin_amdgcn_wmma_f32_16x16x32_bf16(false, A1, false, Bf,
                                                          (short)0, acc[1][t], false, false);
    }
  }

#pragma unroll
  for (int mi = 0; mi < 2; ++mi)
#pragma unroll
    for (int t = 0; t < 4; ++t) {
      const int col = n_band * 64 + t * 16 + l15;
#pragma unroll
      for (int v = 0; v < 8; ++v) {
        const int row = m0 + mi * 16 + v + 8 * hl;
        out[(size_t)row * DMODEL + col] = acc[mi][t][v];
      }
    }
}

extern "C" void kernel_launch(void* const* d_in, const int* in_sizes, int n_in,
                              void* d_out, int out_size, void* d_ws, size_t ws_size,
                              hipStream_t stream) {
  const float*         x    = (const float*)d_in[0];
  const unsigned char* mask = (const unsigned char*)d_in[1];  // jnp.bool_ -> 1 byte
  const float*         wqkv = (const float*)d_in[2];
  const float*         wout = (const float*)d_in[3];
  float* out = (float*)d_out;

  char* ws = (char*)d_ws;
  size_t off = 0;
  auto carve = [&](size_t elems) {
    __bf16* p = (__bf16*)(ws + off);
    off += elems * sizeof(__bf16);
    return p;
  };
  __bf16* xb    = carve((size_t)BATCH * SLEN * DMODEL);           // 16 MB
  __bf16* wqkvb = carve((size_t)3 * DMODEL * DMODEL);             //  6 MB
  __bf16* woutb = carve((size_t)DMODEL * DMODEL);                 //  2 MB
  __bf16* Qb    = carve((size_t)BATCH * NHEAD * SLEN * HDIM);     // 16 MB
  __bf16* Kb    = carve((size_t)BATCH * NHEAD * SLEN * HDIM);     // 16 MB
  __bf16* Vtb   = carve((size_t)BATCH * NHEAD * SLEN * HDIM);     // 16 MB
  __bf16* valsb = carve((size_t)BATCH * SLEN * DMODEL);           // 16 MB
  (void)ws_size; (void)in_sizes; (void)n_in; (void)out_size;

  const int nx = BATCH * SLEN * DMODEL;
  const int nw = 3 * DMODEL * DMODEL;
  const int no = DMODEL * DMODEL;
  convert_bf16<<<(nx + 255) / 256, 256, 0, stream>>>(x, xb, nx);
  convert_bf16<<<(nw + 255) / 256, 256, 0, stream>>>(wqkv, wqkvb, nw);
  convert_bf16<<<(no + 255) / 256, 256, 0, stream>>>(wout, woutb, no);

  qkv_rope  <<<dim3(48, 32), 256, 0, stream>>>(xb, wqkvb, Qb, Kb, Vtb);
  flash_attn<<<dim3(64, 16), 256, 0, stream>>>(Qb, Kb, Vtb, mask, valsb);
  out_proj  <<<dim3(16, 32), 256, 0, stream>>>(valsb, woutb, out);
}
